// EncoderProposals_17454747091625
// MI455X (gfx1250) — matrix operand order repaired and checked by reading, not compiled, not run on hardware
//
#include <hip/hip_runtime.h>
#include <hip/hip_bf16.h>
#include <math.h>

// ---------------------------------------------------------------------------
// EncoderProposals on MI455X (gfx1250): bf16 WMMA pipeline, memory-bound.
// - Weights pre-repacked (once per launch) into WMMA B-fragment layout:
//   each B fragment is one contiguous 32B/lane load (2x global_load_b128).
// - Encoder GEMM+LN and all heads fused into ONE kernel: the LN'd tile goes
//   to global fp32 (for the final gather) and simultaneously into a per-wave
//   LDS bf16 tile that feeds the heads' A fragments -> out_mem is never
//   re-read from HBM (~32% traffic cut vs split kernels).
// ---------------------------------------------------------------------------

typedef __attribute__((ext_vector_type(16))) __bf16 v16bf;
typedef __attribute__((ext_vector_type(8)))  float  v8f;

static constexpr int  CDIM   = 256;
static constexpr int  NCLS   = 91;
static constexpr int  BATCH  = 8;
static constexpr int  HN     = 128;
static constexpr int  WN     = 128;
static constexpr int  SDIM   = HN * WN;            // 16384
static constexpr long RTOT   = (long)BATCH * SDIM; // 131072 rows
static constexpr int  TOPK   = 900;
static constexpr float STRIDEF   = 8.0f;
static constexpr float MIN_SIZE  = 50.0f;
static constexpr float MAX_RATIO = 4.135166556742356f;  // |ln(0.016)|
static constexpr float IMG_WH    = 1024.0f;             // 128*8

// n-tile counts per weight matrix
static constexpr int NT_ENC = 16;  // 256 cols
static constexpr int NT_CLS = 6;   // 91 -> 96 cols
static constexpr int NT_MLP = 16;  // 256 cols
static constexpr int NT_W3  = 1;   // 4 -> 16 cols
static constexpr int KSTEPS = 8;   // 256 / 32

__device__ __forceinline__ v8f wmma_bf16(v16bf a, v16bf b, v8f c) {
  // D = A(16x32 bf16) * B(32x16 bf16) + C(16x16 f32)
  return __builtin_amdgcn_wmma_f32_16x16x32_bf16(
      /*neg_a=*/false, a, /*neg_b=*/false, b,
      /*c_mod=*/(short)0, c, /*reuse_a=*/false, /*reuse_b=*/false);
}

// A fragment (16x32) from row-major fp32, converted to bf16.
// Lane l: row = l&15 ; lanes<16 hold K {0..7,16..23}, lanes>=16 hold {8..15,24..31}.
__device__ __forceinline__ v16bf load_a_frag_f32(const float* __restrict__ A,
                                                 int lda, int K0, int lane) {
  int r  = lane & 15;
  int kb = (lane & 16) ? 8 : 0;
  const float* p = A + (size_t)r * lda + K0 + kb;
  float4 x0 = *reinterpret_cast<const float4*>(p);
  float4 x1 = *reinterpret_cast<const float4*>(p + 4);
  float4 y0 = *reinterpret_cast<const float4*>(p + 16);
  float4 y1 = *reinterpret_cast<const float4*>(p + 20);
  v16bf a;
  a[0]=(__bf16)x0.x; a[1]=(__bf16)x0.y; a[2]=(__bf16)x0.z; a[3]=(__bf16)x0.w;
  a[4]=(__bf16)x1.x; a[5]=(__bf16)x1.y; a[6]=(__bf16)x1.z; a[7]=(__bf16)x1.w;
  a[8]=(__bf16)y0.x; a[9]=(__bf16)y0.y; a[10]=(__bf16)y0.z; a[11]=(__bf16)y0.w;
  a[12]=(__bf16)y1.x; a[13]=(__bf16)y1.y; a[14]=(__bf16)y1.z; a[15]=(__bf16)y1.w;
  return a;
}

// A fragment from a bf16 LDS tile (row-major, stride lda elems).
__device__ __forceinline__ v16bf load_a_frag_bf16(const __bf16* A, int lda,
                                                  int K0, int lane) {
  int r  = lane & 15;
  int kb = (lane & 16) ? 8 : 0;
  const __bf16* p = A + r * lda + K0 + kb;
  v16bf a;
#pragma unroll
  for (int v = 0; v < 4; ++v) { a[2*v] = p[2*v]; a[2*v+1] = p[2*v+1]; }
#pragma unroll
  for (int v = 0; v < 4; ++v) { a[8+2*v] = p[16+2*v]; a[9+2*v] = p[16+2*v+1]; }
  return a;
}

// B fragment (32x16) from row-major fp32 (ldb columns), guarded against ncols.
// Used only by the one-shot repack kernel.
__device__ __forceinline__ v16bf load_b_frag_f32(const float* __restrict__ B,
                                                 int ldb, int ncols,
                                                 int K0, int n0, int lane) {
  int c   = lane & 15;
  int kb  = (lane & 16) ? 8 : 0;
  int col = n0 + c;
  v16bf b;
  if (col < ncols) {
    const float* p = B + (size_t)(K0 + kb) * ldb + col;
#pragma unroll
    for (int v = 0; v < 8; ++v) {
      b[2*v]   = (__bf16)p[(2*v + (v < 4 ? 0 : 8)) * ldb];
      b[2*v+1] = (__bf16)p[(2*v+1 + (v < 4 ? 0 : 8)) * ldb];
    }
  } else {
#pragma unroll
    for (int i = 0; i < 16; ++i) b[i] = (__bf16)0.0f;
  }
  return b;
}

// Packed B fragment: [tile = k*ntiles+n][lane][16 bf16] -> one 32B/lane load.
__device__ __forceinline__ v16bf load_b_frag_packed(const __bf16* __restrict__ P,
                                                    int ntiles, int k, int n,
                                                    int lane) {
  return reinterpret_cast<const v16bf*>(P)[(k * ntiles + n) * 32 + lane];
}

// ---------------------------------------------------------------------------
// Kernel 0: repack one fp32 weight matrix into bf16 B-fragment layout.
// grid = KSTEPS*ntiles blocks x 32 threads. Tiny, runs once per launch.
// ---------------------------------------------------------------------------
__global__ void __launch_bounds__(32)
repack_b_kernel(const float* __restrict__ W, int ldb, int ncols, int ntiles,
                __bf16* __restrict__ dst) {
  const int tile = blockIdx.x;            // k*ntiles + n
  const int k = tile / ntiles;
  const int n = tile % ntiles;
  const int lane = threadIdx.x;
  v16bf b = load_b_frag_f32(W, ldb, ncols, k * 32, n * 16, lane);
  reinterpret_cast<v16bf*>(dst)[tile * 32 + lane] = b;
}

// ---------------------------------------------------------------------------
// Fused kernel: per wave32, one 16-row tile does:
//   1) out_mem = mask0( LN( src @ W_enc + b_enc ) )  (128 f32 accum VGPRs,
//      LN via shfl_xor over 16-lane halves)  -> global fp32 + LDS bf16 tile
//   2) scores = max(out_mem @ W_cls + b_cls)
//   3) h1 = relu(out_mem@w1+b1) -> LDS ; h2 = relu(h1@w2+b2) -> LDS
//   4) deltas = h2 @ w3 + b3 ; box decode ; validity-masked score
// 4 waves / block (LDS: 4 x 8KB bf16 tiles + epsilon).
// ---------------------------------------------------------------------------
__global__ void __launch_bounds__(128)
fused_kernel(const float* __restrict__ src, const unsigned char* __restrict__ mask,
             const __bf16* __restrict__ pW_enc, const float* __restrict__ b_enc,
             const float* __restrict__ ln_g, const float* __restrict__ ln_b,
             const __bf16* __restrict__ pW_cls, const float* __restrict__ b_cls,
             const __bf16* __restrict__ pW1, const float* __restrict__ b1,
             const __bf16* __restrict__ pW2, const float* __restrict__ b2,
             const __bf16* __restrict__ pW3, const float* __restrict__ b3,
             float* __restrict__ out_mem,
             float* __restrict__ scoresW, float* __restrict__ boxesW) {
  __shared__ __bf16 sH[4][16 * CDIM];      // 32 KB, reused out_mem->h1->h2
  __shared__ float  sRowMax[4][16];
  __shared__ float  sDelta[4][16][4];

  const int lane = threadIdx.x & 31;
  const int wave = threadIdx.x >> 5;
  const long rowBase = ((long)blockIdx.x * 4 + wave) * 16;
  const int colh = lane & 15;
  const int rb   = (lane < 16) ? 0 : 8;

  // ======== Phase 1: encoder GEMM + bias + LayerNorm ========================
  {
    const float* A = src + rowBase * CDIM;
    v8f acc[NT_ENC];
#pragma unroll
    for (int n = 0; n < NT_ENC; ++n) acc[n] = (v8f){0,0,0,0,0,0,0,0};

#pragma unroll
    for (int k = 0; k < KSTEPS; ++k) {
      v16bf af = load_a_frag_f32(A, CDIM, k * 32, lane);
#pragma unroll
      for (int n = 0; n < NT_ENC; ++n)
        acc[n] = wmma_bf16(af, load_b_frag_packed(pW_enc, NT_ENC, k, n, lane), acc[n]);
    }

    float gbuf[NT_ENC], bbuf[NT_ENC];
#pragma unroll
    for (int n = 0; n < NT_ENC; ++n) {
      int col = n * 16 + colh;
      gbuf[n] = ln_g[col];
      bbuf[n] = ln_b[col];
    }

    float rs[8], rq[8];
#pragma unroll
    for (int v = 0; v < 8; ++v) { rs[v] = 0.f; rq[v] = 0.f; }

#pragma unroll
    for (int n = 0; n < NT_ENC; ++n) {
      float bb = b_enc[n * 16 + colh];
#pragma unroll
      for (int v = 0; v < 8; ++v) {
        float x = acc[n][v] + bb;
        acc[n][v] = x;
        rs[v] += x;
        rq[v] += x * x;
      }
    }
    // reduce across the 16 lanes of each half (rows 0-7 lanes 0-15, 8-15 lanes 16-31)
#pragma unroll
    for (int m = 1; m <= 8; m <<= 1) {
#pragma unroll
      for (int v = 0; v < 8; ++v) {
        rs[v] += __shfl_xor(rs[v], m, 32);
        rq[v] += __shfl_xor(rq[v], m, 32);
      }
    }
#pragma unroll
    for (int v = 0; v < 8; ++v) {
      float mu  = rs[v] * (1.0f / 256.0f);
      float var = rq[v] * (1.0f / 256.0f) - mu * mu;
      float inv = rsqrtf(var + 1e-5f);
      long  gr  = rowBase + rb + v;
      float mk  = mask[gr] ? 0.0f : 1.0f;
#pragma unroll
      for (int n = 0; n < NT_ENC; ++n) {
        float y = ((acc[n][v] - mu) * inv * gbuf[n] + bbuf[n]) * mk;
        out_mem[gr * CDIM + n * 16 + colh] = y;                      // for gather
        sH[wave][(rb + v) * CDIM + n * 16 + colh] = (__bf16)y;       // for heads
      }
    }
  }
  __syncthreads();

  // A fragments of out_mem, from LDS, held in 64 VGPRs for all heads.
  v16bf a0[KSTEPS];
#pragma unroll
  for (int k = 0; k < KSTEPS; ++k) a0[k] = load_a_frag_bf16(sH[wave], CDIM, k * 32, lane);
  __syncthreads();

  // ======== Phase 2: classification logits -> per-row max ==================
  float rmax[8];
#pragma unroll
  for (int v = 0; v < 8; ++v) rmax[v] = -1e30f;
#pragma unroll
  for (int n = 0; n < NT_CLS; ++n) {          // 6 tiles cover 96 >= 91 cols
    v8f acc = (v8f){0,0,0,0,0,0,0,0};
#pragma unroll
    for (int k = 0; k < KSTEPS; ++k)
      acc = wmma_bf16(a0[k], load_b_frag_packed(pW_cls, NT_CLS, k, n, lane), acc);
    int col = n * 16 + colh;
    bool cvalid = col < NCLS;
    float bb = cvalid ? b_cls[col] : 0.0f;
#pragma unroll
    for (int v = 0; v < 8; ++v) {
      float x = cvalid ? (acc[v] + bb) : -1e30f;
      rmax[v] = fmaxf(rmax[v], x);
    }
  }
#pragma unroll
  for (int m = 1; m <= 8; m <<= 1)
#pragma unroll
    for (int v = 0; v < 8; ++v) rmax[v] = fmaxf(rmax[v], __shfl_xor(rmax[v], m, 32));
  if ((lane & 15) == 0)
#pragma unroll
    for (int v = 0; v < 8; ++v) sRowMax[wave][rb + v] = rmax[v];

  // ======== Phase 3: h1 = relu(out_mem @ w1 + b1) -> LDS ===================
#pragma unroll
  for (int n = 0; n < NT_MLP; ++n) {
    v8f acc = (v8f){0,0,0,0,0,0,0,0};
#pragma unroll
    for (int k = 0; k < KSTEPS; ++k)
      acc = wmma_bf16(a0[k], load_b_frag_packed(pW1, NT_MLP, k, n, lane), acc);
    float bb = b1[n * 16 + colh];
#pragma unroll
    for (int v = 0; v < 8; ++v)
      sH[wave][(rb + v) * CDIM + n * 16 + colh] = (__bf16)fmaxf(acc[v] + bb, 0.0f);
  }
  __syncthreads();

  v16bf a1[KSTEPS];
#pragma unroll
  for (int k = 0; k < KSTEPS; ++k) a1[k] = load_a_frag_bf16(sH[wave], CDIM, k * 32, lane);
  __syncthreads();

  // ======== Phase 4: h2 = relu(h1 @ w2 + b2) -> LDS ========================
#pragma unroll
  for (int n = 0; n < NT_MLP; ++n) {
    v8f acc = (v8f){0,0,0,0,0,0,0,0};
#pragma unroll
    for (int k = 0; k < KSTEPS; ++k)
      acc = wmma_bf16(a1[k], load_b_frag_packed(pW2, NT_MLP, k, n, lane), acc);
    float bb = b2[n * 16 + colh];
#pragma unroll
    for (int v = 0; v < 8; ++v)
      sH[wave][(rb + v) * CDIM + n * 16 + colh] = (__bf16)fmaxf(acc[v] + bb, 0.0f);
  }
  __syncthreads();

  v16bf a2[KSTEPS];
#pragma unroll
  for (int k = 0; k < KSTEPS; ++k) a2[k] = load_a_frag_bf16(sH[wave], CDIM, k * 32, lane);

  // ======== Phase 5: deltas = h2 @ w3 + b3 (4 cols zero-padded to 16) ======
  v8f accd = (v8f){0,0,0,0,0,0,0,0};
#pragma unroll
  for (int k = 0; k < KSTEPS; ++k)
    accd = wmma_bf16(a2[k], load_b_frag_packed(pW3, NT_W3, k, 0, lane), accd);
  if (colh < 4)
#pragma unroll
    for (int v = 0; v < 8; ++v)
      sDelta[wave][rb + v][colh] = accd[v] + b3[colh];
  __syncthreads();

  // ======== Phase 6: per-row box decode + score masking ====================
  if (lane < 16) {
    long gr = rowBase + lane;
    int  s  = (int)(gr % SDIM);
    float cx = ((float)(s % WN) + 0.5f) * STRIDEF;
    float cy = ((float)(s / WN) + 0.5f) * STRIDEF;
    const float pwh = 4.0f * STRIDEF;
    float dx = sDelta[wave][lane][0];
    float dy = sDelta[wave][lane][1];
    float dw = fminf(fmaxf(sDelta[wave][lane][2], -MAX_RATIO), MAX_RATIO);
    float dh = fminf(fmaxf(sDelta[wave][lane][3], -MAX_RATIO), MAX_RATIO);
    float gx = cx + pwh * dx, gy = cy + pwh * dy;
    float gw = pwh * expf(dw), gh = pwh * expf(dh);
    float x1 = fminf(fmaxf(gx - 0.5f * gw, 0.0f), IMG_WH);
    float y1 = fminf(fmaxf(gy - 0.5f * gh, 0.0f), IMG_WH);
    float x2 = fminf(fmaxf(gx + 0.5f * gw, 0.0f), IMG_WH);
    float y2 = fminf(fmaxf(gy + 0.5f * gh, 0.0f), IMG_WH);
    boxesW[gr * 4 + 0] = x1; boxesW[gr * 4 + 1] = y1;
    boxesW[gr * 4 + 2] = x2; boxesW[gr * 4 + 3] = y2;
    bool valid = (x2 - x1 >= MIN_SIZE) && (y2 - y1 >= MIN_SIZE) && !mask[gr];
    scoresW[gr] = valid ? sRowMax[wave][lane] : -1e9f;
  }
}

// ---------------------------------------------------------------------------
// Kernel 3: per-batch top-900 via 11-bit radix select on ordered float keys.
// ---------------------------------------------------------------------------
__device__ __forceinline__ unsigned f2ord(float f) {
  unsigned u = __float_as_uint(f);
  return (u & 0x80000000u) ? ~u : (u | 0x80000000u);
}

__global__ void __launch_bounds__(1024)
topk_kernel(const float* __restrict__ scoresW, int* __restrict__ selIdx) {
  __shared__ unsigned hist[2048];
  __shared__ unsigned sThresh, sAbove, cA, cT;
  const int b = blockIdx.x;
  const float* sc = scoresW + (long)b * SDIM;

  for (int i = threadIdx.x; i < 2048; i += 1024) hist[i] = 0;
  __syncthreads();
  for (int i = threadIdx.x; i < SDIM; i += 1024)
    atomicAdd(&hist[f2ord(sc[i]) >> 21], 1u);
  __syncthreads();
  if (threadIdx.x == 0) {
    unsigned cum = 0;
    int bin = 2047;
    for (; bin > 0; --bin) {
      unsigned c = hist[bin];
      if (cum + c >= (unsigned)TOPK) break;
      cum += c;
    }
    sThresh = (unsigned)bin; sAbove = cum; cA = 0; cT = 0;
  }
  __syncthreads();
  const unsigned T = sThresh, above = sAbove;
  for (int i = threadIdx.x; i < SDIM; i += 1024) {
    unsigned bin = f2ord(sc[i]) >> 21;
    if (bin > T) {
      unsigned slot = atomicAdd(&cA, 1u);
      selIdx[b * TOPK + slot] = i;
    } else if (bin == T) {
      unsigned slot = above + atomicAdd(&cT, 1u);
      if (slot < (unsigned)TOPK) selIdx[b * TOPK + slot] = i;
    }
  }
}

// ---------------------------------------------------------------------------
// Kernel 4: gather [feats(256) | box(4) | score(1)] -> out (8,900,261).
// ---------------------------------------------------------------------------
__global__ void __launch_bounds__(256)
gather_kernel(const float* __restrict__ out_mem, const float* __restrict__ boxesW,
              const float* __restrict__ scoresW, const int* __restrict__ selIdx,
              float* __restrict__ out) {
  const int r = blockIdx.x;             // 0..B*TOPK-1
  const int b = r / TOPK;
  const long src_row = (long)b * SDIM + selIdx[r];
  float* o = out + (long)r * 261;
  const int t = threadIdx.x;
  o[t] = out_mem[src_row * CDIM + t];
  if (t < 4)  o[256 + t] = boxesW[src_row * 4 + t];
  if (t == 0) o[260] = scoresW[src_row];
}

// ---------------------------------------------------------------------------
extern "C" void kernel_launch(void* const* d_in, const int* in_sizes, int n_in,
                              void* d_out, int out_size, void* d_ws, size_t ws_size,
                              hipStream_t stream) {
  const float*         src   = (const float*)d_in[0];
  const unsigned char* mask  = (const unsigned char*)d_in[1];
  const float*         W_enc = (const float*)d_in[2];
  const float*         b_enc = (const float*)d_in[3];
  const float*         ln_g  = (const float*)d_in[4];
  const float*         ln_b  = (const float*)d_in[5];
  const float*         W_cls = (const float*)d_in[6];
  const float*         b_cls = (const float*)d_in[7];
  const float*         w1    = (const float*)d_in[8];
  const float*         b1    = (const float*)d_in[9];
  const float*         w2    = (const float*)d_in[10];
  const float*         b2    = (const float*)d_in[11];
  const float*         w3    = (const float*)d_in[12];
  const float*         b3    = (const float*)d_in[13];
  float* out = (float*)d_out;

  // Workspace layout (~137.5 MB)
  float* out_mem = (float*)d_ws;
  float* scoresW = out_mem + (size_t)RTOT * CDIM;
  float* boxesW  = scoresW + RTOT;
  int*   selIdx  = (int*)(boxesW + (size_t)RTOT * 4);
  // packed bf16 weights (align to 32B past selIdx)
  size_t selEnd = ((size_t)(selIdx + BATCH * TOPK) + 31) & ~(size_t)31;
  __bf16* pW_enc = (__bf16*)selEnd;
  __bf16* pW_cls = pW_enc + (size_t)KSTEPS * NT_ENC * 32 * 16;
  __bf16* pW1    = pW_cls + (size_t)KSTEPS * NT_CLS * 32 * 16;
  __bf16* pW2    = pW1    + (size_t)KSTEPS * NT_MLP * 32 * 16;
  __bf16* pW3    = pW2    + (size_t)KSTEPS * NT_MLP * 32 * 16;

  // --- one-shot weight repack into WMMA B-fragment layout -------------------
  repack_b_kernel<<<KSTEPS * NT_ENC, 32, 0, stream>>>(W_enc, CDIM, CDIM, NT_ENC, pW_enc);
  repack_b_kernel<<<KSTEPS * NT_CLS, 32, 0, stream>>>(W_cls, NCLS, NCLS, NT_CLS, pW_cls);
  repack_b_kernel<<<KSTEPS * NT_MLP, 32, 0, stream>>>(w1, CDIM, CDIM, NT_MLP, pW1);
  repack_b_kernel<<<KSTEPS * NT_MLP, 32, 0, stream>>>(w2, CDIM, CDIM, NT_MLP, pW2);
  repack_b_kernel<<<KSTEPS * NT_W3,  32, 0, stream>>>(w3, 4, 4, NT_W3, pW3);

  // --- fused main pipeline --------------------------------------------------
  fused_kernel<<<(int)(RTOT / 64), 128, 0, stream>>>(
      src, mask, pW_enc, b_enc, ln_g, ln_b,
      pW_cls, b_cls, pW1, b1, pW2, b2, pW3, b3,
      out_mem, scoresW, boxesW);
  topk_kernel<<<BATCH, 1024, 0, stream>>>(scoresW, selIdx);
  gather_kernel<<<BATCH * TOPK, 256, 0, stream>>>(
      out_mem, boxesW, scoresW, selIdx, out);
}
